// GNN_60017872994330
// MI455X (gfx1250) — compile-verified, matrix operand order checked by reading, not run on hardware
//
#include <hip/hip_runtime.h>
#include <math.h>
#include <stdint.h>

// ---------------------------------------------------------------------------
// CDNA5 (gfx1250) types for WMMA
// ---------------------------------------------------------------------------
typedef __attribute__((ext_vector_type(16))) __bf16 v16bf;
typedef __attribute__((ext_vector_type(8)))  float  v8f;

union FragAB {
    v16bf v;
    uint4 u[2];
};

__device__ __forceinline__ unsigned short f2bf(float f) {
    unsigned int u = __float_as_uint(f);
    unsigned int r = u + 0x7FFFu + ((u >> 16) & 1u);   // round-to-nearest-even
    return (unsigned short)(r >> 16);
}

__device__ __forceinline__ void atomicMaxF(float* addr, float val) {
    if (val >= 0.0f)
        atomicMax((int*)addr, __float_as_int(val));
    else
        atomicMin((unsigned int*)addr, (unsigned int)__float_as_int(val));
}

// ---------------------------------------------------------------------------
// CDNA5 async DMA: global -> LDS, tracked by ASYNCcnt (bypasses VGPRs).
// LDS operand = wave-relative LDS byte offset = low 32 bits of flat address.
// ---------------------------------------------------------------------------
__device__ __forceinline__ void async_b128_to_lds(void* lds_ptr, const void* gptr) {
    unsigned ldsoff = (unsigned)(unsigned long long)(uintptr_t)lds_ptr;
    asm volatile("global_load_async_to_lds_b128 %0, %1, off"
                 :: "v"(ldsoff), "v"((unsigned long long)(uintptr_t)gptr)
                 : "memory");
}
__device__ __forceinline__ void wait_asynccnt0() {
    asm volatile("s_wait_asynccnt 0x0" ::: "memory");
}

// ---------------------------------------------------------------------------
// Utility kernels
// ---------------------------------------------------------------------------
__global__ __launch_bounds__(256)
void fill_f32(float* __restrict__ p, float v, long n) {
    long i = (long)blockIdx.x * blockDim.x + threadIdx.x;
    if (i < n) p[i] = v;
}

__global__ __launch_bounds__(256)
void cvt_bf16(const float* __restrict__ x, unsigned short* __restrict__ y, long n) {
    long i = (long)blockIdx.x * blockDim.x + threadIdx.x;
    if (i < n) y[i] = f2bf(x[i]);
}

// W [256, Ncol] f32 row-major  ->  Wt [Ncol, 256] bf16 row-major
__global__ __launch_bounds__(256)
void wtrans_bf16(const float* __restrict__ W, unsigned short* __restrict__ Wt,
                 int Ncol, int total) {
    int i = blockIdx.x * blockDim.x + threadIdx.x;
    if (i >= total) return;
    int k = i / Ncol;
    int n = i - k * Ncol;
    Wt[(size_t)n * 256 + k] = f2bf(W[i]);
}

// ---------------------------------------------------------------------------
// A-tile staging, shared by both GEMM kernels:
// 16 rows x 256 K bf16 (8 KB) via async DMA into LDS.
// ---------------------------------------------------------------------------
__device__ __forceinline__ void stage_A_tile(uint4* ldsA, const unsigned short* A,
                                             int mtile) {
    const uint4* Au4 = (const uint4*)A;                     // row pitch = 32 uint4
    for (int c = threadIdx.x; c < 512; c += 256) {
        int r = c >> 5;                                     // row in tile
        int u = c & 31;                                     // uint4 in row
        async_b128_to_lds(&ldsA[c], &Au4[(size_t)(mtile * 16 + r) * 32 + u]);
    }
    wait_asynccnt0();
    __syncthreads();
}

// ---------------------------------------------------------------------------
// Hot GEMM:  C[M,256] = A[M,256](bf16) * Wt[256,256]^T(bf16) + bias
// Ncol fixed at 256: 8 waves x 2 tiles = 16 N-tiles, branch-free K-loop
// with two independent WMMA accumulator chains sharing the A fragment.
// ---------------------------------------------------------------------------
__global__ __launch_bounds__(256)
void gemm256_bf16_wmma(const unsigned short* __restrict__ A,   // [M,256] bf16
                       const unsigned short* __restrict__ Wt,  // [256,256] bf16
                       const float* __restrict__ bias,         // [256]
                       float* __restrict__ C) {                // [M,256] f32
    __shared__ uint4 ldsA[512];
    const int mtile = blockIdx.x;
    stage_A_tile(ldsA, A, mtile);

    const int wave  = __builtin_amdgcn_readfirstlane((int)(threadIdx.x >> 5));
    const int lane  = threadIdx.x & 31;
    const int mloc  = lane & 15;                            // row (A) / col (B)
    const int khalf = (lane >> 4);                          // 0 or 1 (uint4 units)
    const int col0  = wave * 32 + mloc;                     // two 16-wide tiles/wave
    const uint4* wrow0 = (const uint4*)(Wt + (size_t)col0 * 256);
    const uint4* wrow1 = wrow0 + 16 * 32;                   // +16 columns

    v8f acc0 = {0.f, 0.f, 0.f, 0.f, 0.f, 0.f, 0.f, 0.f};
    v8f acc1 = {0.f, 0.f, 0.f, 0.f, 0.f, 0.f, 0.f, 0.f};

#pragma unroll
    for (int kt = 0; kt < 32; kt += 4) {                    // K step = 32 bf16 = 4 uint4
        const int kb = kt + khalf;
        FragAB a, b0, b1;
        a.u[0]  = ldsA[mloc * 32 + kb];
        a.u[1]  = ldsA[mloc * 32 + kb + 2];
        b0.u[0] = wrow0[kb];
        b0.u[1] = wrow0[kb + 2];
        b1.u[0] = wrow1[kb];
        b1.u[1] = wrow1[kb + 2];
        __builtin_prefetch(wrow0 + kb + 4, 0, 1);           // global_prefetch_b8
        acc0 = __builtin_amdgcn_wmma_f32_16x16x32_bf16(
            false, a.v, false, b0.v, (short)0, acc0, false, false);
        acc1 = __builtin_amdgcn_wmma_f32_16x16x32_bf16(
            false, a.v, false, b1.v, (short)0, acc1, false, false);
    }

    const int mbase = mtile * 16 + (khalf << 3);            // +8 for upper half-wave
    const float bn0 = bias[col0];
    const float bn1 = bias[col0 + 16];
#pragma unroll
    for (int r = 0; r < 8; ++r) {
        C[(size_t)(mbase + r) * 256 + col0]      = acc0[r] + bn0;
        C[(size_t)(mbase + r) * 256 + col0 + 16] = acc1[r] + bn1;
    }
}

// ---------------------------------------------------------------------------
// Small GEMM (final head, Ncol = 64 or 16): one 16x16 tile per wave.
// Idle-wave guard is a *scalar* branch (readfirstlane), EXEC never touched.
// ---------------------------------------------------------------------------
__global__ __launch_bounds__(256)
void gemm_small_bf16_wmma(const unsigned short* __restrict__ A,
                          const unsigned short* __restrict__ Wt,
                          const float* __restrict__ bias,
                          float* __restrict__ C,
                          int Ncol, int Ntiles) {
    __shared__ uint4 ldsA[512];
    const int mtile = blockIdx.x;
    stage_A_tile(ldsA, A, mtile);

    const int wave = __builtin_amdgcn_readfirstlane((int)(threadIdx.x >> 5));
    if (wave >= Ntiles) return;                             // uniform scalar branch

    const int lane  = threadIdx.x & 31;
    const int mloc  = lane & 15;
    const int khalf = (lane >> 4);
    const int col   = wave * 16 + mloc;
    const uint4* wrow = (const uint4*)(Wt + (size_t)col * 256);

    v8f acc = {0.f, 0.f, 0.f, 0.f, 0.f, 0.f, 0.f, 0.f};
#pragma unroll
    for (int kt = 0; kt < 32; kt += 4) {
        const int kb = kt + khalf;
        FragAB a, b;
        a.u[0] = ldsA[mloc * 32 + kb];
        a.u[1] = ldsA[mloc * 32 + kb + 2];
        b.u[0] = wrow[kb];
        b.u[1] = wrow[kb + 2];
        acc = __builtin_amdgcn_wmma_f32_16x16x32_bf16(
            false, a.v, false, b.v, (short)0, acc, false, false);
    }

    const int   mbase = mtile * 16 + (khalf << 3);
    const float bn    = bias[col];
#pragma unroll
    for (int r = 0; r < 8; ++r)
        C[(size_t)(mbase + r) * Ncol + col] = acc[r] + bn;
}

// ---------------------------------------------------------------------------
// Edge kernels (4 heads always; CH = 64 for layers, 16 for final head)
// ---------------------------------------------------------------------------
template <int CH>
__global__ __launch_bounds__(256)
void edge_score(const float* __restrict__ q, const float* __restrict__ k,
                const int* __restrict__ src, const int* __restrict__ dst,
                float* __restrict__ score, float* __restrict__ mbuf,
                int E, float scale) {
    int idx = blockIdx.x * blockDim.x + threadIdx.x;
    int e = idx >> 2, h = idx & 3;
    if (e >= E) return;
    int s = src[e], d = dst[e];
    const float4* qp = (const float4*)(q + (size_t)d * (4 * CH) + h * CH);
    const float4* kp = (const float4*)(k + (size_t)s * (4 * CH) + h * CH);
    float acc = 0.f;
#pragma unroll
    for (int i = 0; i < CH / 4; ++i) {
        float4 a = qp[i], b = kp[i];
        acc += a.x * b.x + a.y * b.y + a.z * b.z + a.w * b.w;
    }
    acc *= scale;
    score[(size_t)e * 4 + h] = acc;
    atomicMaxF(mbuf + (size_t)d * 4 + h, acc);
}

__global__ __launch_bounds__(256)
void edge_exp(const int* __restrict__ dst, float* __restrict__ score,
              const float* __restrict__ mbuf, float* __restrict__ sbuf, int E) {
    int idx = blockIdx.x * blockDim.x + threadIdx.x;
    int e = idx >> 2, h = idx & 3;
    if (e >= E) return;
    int d = dst[e];
    float p = __expf(score[(size_t)e * 4 + h] - mbuf[(size_t)d * 4 + h]);
    score[(size_t)e * 4 + h] = p;
    atomicAdd(sbuf + (size_t)d * 4 + h, p);
}

// One thread per (edge, feature-channel): coalesced v[src] gather + one
// global_atomic_add_f32 per thread.  PITCH = heads*ch (256 or 64).
template <int PITCH>
__global__ __launch_bounds__(256)
void edge_agg(const int* __restrict__ src, const int* __restrict__ dst,
              const float* __restrict__ score, const float* __restrict__ sbuf,
              const float* __restrict__ v, float* __restrict__ agg, int E) {
    long idx = (long)blockIdx.x * blockDim.x + threadIdx.x;
    int e = (int)(idx / PITCH);
    int c = (int)(idx - (long)e * PITCH);
    if (e >= E) return;
    int h = c / (PITCH / 4);
    int s = src[e], d = dst[e];
    float den   = sbuf[(size_t)d * 4 + h];
    float alpha = score[(size_t)e * 4 + h] / (den == 0.f ? 1.f : den);
    atomicAdd(agg + (size_t)d * PITCH + c, alpha * v[(size_t)s * PITCH + c]);
}

// ---------------------------------------------------------------------------
// Fused ELU + LayerNorm over 256 features; one wave32 per row.
// agg already contains (scatter-sum + h@Ws + bs); lin = h@linW + linB.
// ---------------------------------------------------------------------------
__global__ __launch_bounds__(256)
void fuse_elu_ln(const float* __restrict__ agg, const float* __restrict__ lin,
                 const float* __restrict__ g, const float* __restrict__ b,
                 float* __restrict__ hout, int N) {
    int wid  = (int)((blockIdx.x * (long)blockDim.x + threadIdx.x) >> 5);
    int lane = threadIdx.x & 31;
    if (wid >= N) return;

    const float4* ar = (const float4*)(agg + (size_t)wid * 256);
    const float4* lr = (const float4*)(lin + (size_t)wid * 256);
    float4 a0 = ar[lane * 2],     a1 = ar[lane * 2 + 1];
    float4 l0 = lr[lane * 2],     l1 = lr[lane * 2 + 1];

    float vals[8] = {a0.x + l0.x, a0.y + l0.y, a0.z + l0.z, a0.w + l0.w,
                     a1.x + l1.x, a1.y + l1.y, a1.z + l1.z, a1.w + l1.w};
    float sum = 0.f;
#pragma unroll
    for (int j = 0; j < 8; ++j) {
        float x = vals[j];
        x = (x > 0.f) ? x : (__expf(x) - 1.f);              // ELU
        vals[j] = x;
        sum += x;
    }
#pragma unroll
    for (int off = 16; off; off >>= 1) sum += __shfl_xor(sum, off, 32);
    float mu = sum * (1.f / 256.f);

    float vs = 0.f;
#pragma unroll
    for (int j = 0; j < 8; ++j) { float d = vals[j] - mu; vs += d * d; }
#pragma unroll
    for (int off = 16; off; off >>= 1) vs += __shfl_xor(vs, off, 32);
    float inv = rsqrtf(vs * (1.f / 256.f) + 1e-5f);

    float* hr = hout + (size_t)wid * 256;
#pragma unroll
    for (int j = 0; j < 8; ++j) {
        int col = lane * 8 + j;
        hr[col] = (vals[j] - mu) * inv * g[col] + b[col];
    }
}

// ---------------------------------------------------------------------------
// Final head: mean over 4 heads + skip + linear, LayerNorm over 16 classes.
// 16 lanes per row (2 rows per wave32).
// ---------------------------------------------------------------------------
__global__ __launch_bounds__(256)
void final_out(const float* __restrict__ aggH, const float* __restrict__ sf,
               const float* __restrict__ linf,
               const float* __restrict__ g, const float* __restrict__ b,
               float* __restrict__ out, int N) {
    long gid = (long)blockIdx.x * blockDim.x + threadIdx.x;
    int row = (int)(gid >> 4);
    int c   = (int)(gid & 15);
    if (row >= N) return;

    const float* ah = aggH + (size_t)row * 64;
    float v = 0.25f * (ah[c] + ah[16 + c] + ah[32 + c] + ah[48 + c])
            + sf[(size_t)row * 16 + c] + linf[(size_t)row * 16 + c];

    float sum = v;
#pragma unroll
    for (int off = 8; off; off >>= 1) sum += __shfl_xor(sum, off, 32);
    float mu = sum * (1.f / 16.f);
    float d  = v - mu;
    float vs = d * d;
#pragma unroll
    for (int off = 8; off; off >>= 1) vs += __shfl_xor(vs, off, 32);
    out[(size_t)row * 16 + c] = d * rsqrtf(vs * (1.f / 16.f) + 1e-5f) * g[c] + b[c];
}

// ---------------------------------------------------------------------------
// Host orchestration
// ---------------------------------------------------------------------------
extern "C" void kernel_launch(void* const* d_in, const int* in_sizes, int n_in,
                              void* d_out, int out_size, void* d_ws, size_t ws_size,
                              hipStream_t stream) {
    const int D = 256, DEPTH = 3;
    const int N = in_sizes[0] / D;
    const int E = in_sizes[1] / 2;

    const float* x   = (const float*)d_in[0];
    const int*   ei  = (const int*)d_in[1];
    const int*   src = ei;
    const int*   dst = ei + E;
    const float *Wq = (const float*)d_in[2],  *bq = (const float*)d_in[3];
    const float *Wk = (const float*)d_in[4],  *bk = (const float*)d_in[5];
    const float *Wv = (const float*)d_in[6],  *bv = (const float*)d_in[7];
    const float *Ws = (const float*)d_in[8],  *bs = (const float*)d_in[9];
    const float *lnG = (const float*)d_in[10], *lnB = (const float*)d_in[11];
    const float *linW = (const float*)d_in[12], *linB = (const float*)d_in[13];
    const float *fWq = (const float*)d_in[14], *fbq = (const float*)d_in[15];
    const float *fWk = (const float*)d_in[16], *fbk = (const float*)d_in[17];
    const float *fWv = (const float*)d_in[18], *fbv = (const float*)d_in[19];
    const float *fWs = (const float*)d_in[20], *fbs = (const float*)d_in[21];
    const float *flnG = (const float*)d_in[22], *flnB = (const float*)d_in[23];
    const float *flinW = (const float*)d_in[24], *flinB = (const float*)d_in[25];

    // ---- workspace carve-up -------------------------------------------------
    char* ws = (char*)d_ws;
    size_t off = 0;
    auto take = [&](size_t bytes) {
        size_t r = off;
        off += (bytes + 255) & ~(size_t)255;
        return r;
    };
    size_t o_wt   = take((size_t)DEPTH * 5 * 256 * 256 * 2);   // layer weights bf16^T
    size_t o_wtf  = take((size_t)(3 * 64 + 2 * 16) * 256 * 2); // final weights bf16^T
    size_t o_h    = take((size_t)N * D * 4);
    size_t o_hb   = take((size_t)N * D * 2);
    size_t o_q    = take((size_t)N * D * 4);
    size_t o_k    = take((size_t)N * D * 4);
    size_t o_v    = take((size_t)N * D * 4);
    size_t o_lin  = take((size_t)N * D * 4);
    size_t o_agg  = take((size_t)N * D * 4);
    size_t o_sc   = take((size_t)E * 4 * 4);
    size_t o_m    = take((size_t)N * 4 * 4);
    size_t o_s    = take((size_t)N * 4 * 4);
    size_t o_aggH = take((size_t)N * 64 * 4);
    size_t o_sf   = take((size_t)N * 16 * 4);
    size_t o_lf   = take((size_t)N * 16 * 4);
    (void)ws_size; (void)n_in; (void)out_size;

    float*          hbuf = (float*)(ws + o_h);
    unsigned short* hb   = (unsigned short*)(ws + o_hb);
    float* qb = (float*)(ws + o_q);
    float* kb = (float*)(ws + o_k);
    float* vb = (float*)(ws + o_v);
    float* lb = (float*)(ws + o_lin);
    float* ab = (float*)(ws + o_agg);
    float* sc = (float*)(ws + o_sc);
    float* mb = (float*)(ws + o_m);
    float* sb = (float*)(ws + o_s);
    float* aggH = (float*)(ws + o_aggH);
    float* sfb  = (float*)(ws + o_sf);
    float* lfb  = (float*)(ws + o_lf);

    const int Mtiles = N / 16;
    const int WTELEM = 256 * 256;

    // ---- weight conversion / transpose (bf16) -------------------------------
    {
        int blocks = (WTELEM + 255) / 256;
        for (int i = 0; i < DEPTH; ++i) {
            unsigned short* base = (unsigned short*)(ws + o_wt) + (size_t)i * 5 * WTELEM;
            wtrans_bf16<<<blocks, 256, 0, stream>>>(Wq  + (size_t)i * WTELEM, base + 0 * WTELEM, 256, WTELEM);
            wtrans_bf16<<<blocks, 256, 0, stream>>>(Wk  + (size_t)i * WTELEM, base + 1 * WTELEM, 256, WTELEM);
            wtrans_bf16<<<blocks, 256, 0, stream>>>(Wv  + (size_t)i * WTELEM, base + 2 * WTELEM, 256, WTELEM);
            wtrans_bf16<<<blocks, 256, 0, stream>>>(Ws  + (size_t)i * WTELEM, base + 3 * WTELEM, 256, WTELEM);
            wtrans_bf16<<<blocks, 256, 0, stream>>>(linW + (size_t)i * WTELEM, base + 4 * WTELEM, 256, WTELEM);
        }
        unsigned short* fb = (unsigned short*)(ws + o_wtf);
        wtrans_bf16<<<(256 * 64 + 255) / 256, 256, 0, stream>>>(fWq, fb,                 64, 256 * 64);
        wtrans_bf16<<<(256 * 64 + 255) / 256, 256, 0, stream>>>(fWk, fb + 64 * 256,      64, 256 * 64);
        wtrans_bf16<<<(256 * 64 + 255) / 256, 256, 0, stream>>>(fWv, fb + 2 * 64 * 256,  64, 256 * 64);
        wtrans_bf16<<<(256 * 16 + 255) / 256, 256, 0, stream>>>(fWs,   fb + 3 * 64 * 256,            16, 256 * 16);
        wtrans_bf16<<<(256 * 16 + 255) / 256, 256, 0, stream>>>(flinW, fb + 3 * 64 * 256 + 16 * 256, 16, 256 * 16);
    }

    const long nd = (long)N * D;
    const int cvtBlocks  = (int)((nd + 255) / 256);
    const int edgeBlocks = (E * 4 + 255) / 256;
    const int rowWaveBlk = (N + 7) / 8;           // one wave32 per row, 8 waves/block

    // ---- 3 TransformerConv layers ------------------------------------------
    for (int i = 0; i < DEPTH; ++i) {
        const float* cur = (i == 0) ? x : hbuf;
        cvt_bf16<<<cvtBlocks, 256, 0, stream>>>(cur, hb, nd);

        unsigned short* wt = (unsigned short*)(ws + o_wt) + (size_t)i * 5 * WTELEM;
        gemm256_bf16_wmma<<<Mtiles, 256, 0, stream>>>(hb, wt + 0 * WTELEM, bq   + (size_t)i * D, qb);
        gemm256_bf16_wmma<<<Mtiles, 256, 0, stream>>>(hb, wt + 1 * WTELEM, bk   + (size_t)i * D, kb);
        gemm256_bf16_wmma<<<Mtiles, 256, 0, stream>>>(hb, wt + 2 * WTELEM, bv   + (size_t)i * D, vb);
        gemm256_bf16_wmma<<<Mtiles, 256, 0, stream>>>(hb, wt + 3 * WTELEM, bs   + (size_t)i * D, ab); // agg := skip
        gemm256_bf16_wmma<<<Mtiles, 256, 0, stream>>>(hb, wt + 4 * WTELEM, linB + (size_t)i * D, lb);

        fill_f32<<<(N * 4 + 255) / 256, 256, 0, stream>>>(mb, -INFINITY, (long)N * 4);
        fill_f32<<<(N * 4 + 255) / 256, 256, 0, stream>>>(sb, 0.f, (long)N * 4);

        edge_score<64><<<edgeBlocks, 256, 0, stream>>>(qb, kb, src, dst, sc, mb, E, 0.125f); // 1/sqrt(64)
        edge_exp<<<edgeBlocks, 256, 0, stream>>>(dst, sc, mb, sb, E);
        edge_agg<256><<<(int)(((long)E * 256 + 255) / 256), 256, 0, stream>>>(src, dst, sc, sb, vb, ab, E);

        fuse_elu_ln<<<rowWaveBlk, 256, 0, stream>>>(ab, lb, lnG + (size_t)i * D, lnB + (size_t)i * D, hbuf, N);
    }

    // ---- final head (4 heads x 16 classes, concat=False) --------------------
    cvt_bf16<<<cvtBlocks, 256, 0, stream>>>(hbuf, hb, nd);
    unsigned short* fwt = (unsigned short*)(ws + o_wtf);
    gemm_small_bf16_wmma<<<Mtiles, 256, 0, stream>>>(hb, fwt,                fbq, qb, 64, 4);
    gemm_small_bf16_wmma<<<Mtiles, 256, 0, stream>>>(hb, fwt + 64 * 256,     fbk, kb, 64, 4);
    gemm_small_bf16_wmma<<<Mtiles, 256, 0, stream>>>(hb, fwt + 2 * 64 * 256, fbv, vb, 64, 4);
    gemm_small_bf16_wmma<<<Mtiles, 256, 0, stream>>>(hb, fwt + 3 * 64 * 256,            fbs,   sfb, 16, 1);
    gemm_small_bf16_wmma<<<Mtiles, 256, 0, stream>>>(hb, fwt + 3 * 64 * 256 + 16 * 256, flinB, lfb, 16, 1);

    fill_f32<<<(N * 4 + 255) / 256, 256, 0, stream>>>(mb, -INFINITY, (long)N * 4);
    fill_f32<<<(N * 4 + 255) / 256, 256, 0, stream>>>(sb, 0.f, (long)N * 4);
    fill_f32<<<(N * 64 + 255) / 256, 256, 0, stream>>>(aggH, 0.f, (long)N * 64);

    edge_score<16><<<edgeBlocks, 256, 0, stream>>>(qb, kb, src, dst, sc, mb, E, 0.25f);   // 1/sqrt(16)
    edge_exp<<<edgeBlocks, 256, 0, stream>>>(dst, sc, mb, sb, E);
    edge_agg<64><<<(int)(((long)E * 64 + 255) / 256), 256, 0, stream>>>(src, dst, sc, sb, vb, aggH, E);

    final_out<<<(N * 16 + 255) / 256, 256, 0, stream>>>(aggH, sfb, lfb, flnG, flnB,
                                                        (float*)d_out, N);
}
